// LSTMCell_84593675862591
// MI455X (gfx1250) — matrix operand order repaired and checked by reading, not compiled, not run on hardware
//
#include <hip/hip_runtime.h>

// ---------------- problem constants (from reference) ----------------
#define BATCH   512
#define UNITS   2048
#define KSP     2
#define NSP     4
#define NG      (UNITS / NSP)     // 512 columns per gate per shard
#define KD      UNITS             // 2048 = xh feature width per k-shard
#define KTOT    (KSP * KD)        // 4096 total reduction dim

// ---------------- tiling ----------------
#define M_TILE  256               // batch rows per block  (512/256 = 2 row tiles)
#define J_TILE  32                // gate columns per block (x4 gates internally)
#define K_TILE  32                // one v_wmma_f32_16x16x32_bf16 depth
#define THREADS 512               // 16 wave32s; wave w owns rows [16w,16w+16)
#define K_ITERS (KTOT / K_TILE)   // 128

#define A_RS 40                   // LDS row stride in bf16 (32 + 8 pad, 16B-aligned rows)
#define B_RS 40

typedef __attribute__((ext_vector_type(8)))  float   v8f;
typedef __attribute__((ext_vector_type(16))) float   v16f;
typedef __attribute__((ext_vector_type(2)))  float   v2f;
typedef __attribute__((ext_vector_type(16))) __bf16  v16bf;
typedef __attribute__((ext_vector_type(8)))  __bf16  v8bf;
typedef __attribute__((ext_vector_type(2)))  __bf16  v2bf;
typedef __attribute__((ext_vector_type(4)))  float   f4;

__device__ __forceinline__ float fast_sigmoid(float x) {
    return 1.0f / (1.0f + __expf(-x));               // x->-inf: exp->inf -> 0; ok
}
__device__ __forceinline__ float fast_tanh(float x) {
    return 1.0f - 2.0f / (__expf(2.0f * x) + 1.0f);  // saturates cleanly to +/-1
}

__global__ __launch_bounds__(THREADS)
void lstm_cell_wmma(const float* __restrict__ x,   // [512, 2048]
                    const float* __restrict__ hs,  // [2, 512, 1024]
                    const float* __restrict__ cs,  // [4, 512, 512]
                    const float* __restrict__ ws,  // [8, 2048, 2048]
                    float* __restrict__ xout,      // [512, 2048]
                    float* __restrict__ cout)      // [512, 2048]
{
    const int tid    = threadIdx.x;
    const int wave   = tid >> 5;          // 0..15 -> row group
    const int lane   = tid & 31;
    const int lane16 = lane & 15;
    const int hi     = lane >> 4;         // half-wave selector per ISA layouts

    const int b0 = blockIdx.x * M_TILE;   // batch-row tile
    const int j0 = blockIdx.y * J_TILE;   // gate-column tile (within 512)
    const int n  = blockIdx.z;            // n-shard 0..3

    // double-buffered bf16 staging: A = xh tile, B = 4 gate weight tiles (j-major)
    __shared__ __bf16 As[2][M_TILE][A_RS];            // 40 KB
    __shared__ __bf16 Bs[2][4][J_TILE][B_RS];         // 20 KB

    // ---- cooperative-load thread mapping ----
    const int arow  = tid >> 1;           // 0..255
    const int ahalf = tid & 1;            // which 16-float half of the 32-wide K slice
    const int bj    = tid & 31;           // gate column within tile
    const int bk    = (tid >> 5) * 2;     // K-row pair 0,2,..,30

    f4    aReg[4];                        // 16 fp32 of xh
    float bReg[4][2];                     // 2 fp32 per gate

    auto loadA = [&](int kt) {
        const int kk    = kt * K_TILE;
        const int shard = kk >> 11;       // /2048
        const int d     = kk & 2047;
        const float* src;
        if (d < 1024)  // x region of xh
            src = x  + (size_t)(b0 + arow) * UNITS + shard * 1024 + d;
        else           // hs region of xh
            src = hs + ((size_t)shard * BATCH + (b0 + arow)) * 1024 + (d - 1024);
        src += ahalf * 16;
        aReg[0] = *(const f4*)(src + 0);
        aReg[1] = *(const f4*)(src + 4);
        aReg[2] = *(const f4*)(src + 8);
        aReg[3] = *(const f4*)(src + 12);
    };

    auto loadB = [&](int kt) {
        const int kk    = kt * K_TILE;
        const int shard = kk >> 11;
        const int d     = kk & 2047;
        const float* wbase = ws + ((size_t)(shard * NSP + n) << 22)   // *2048*2048
                                + (size_t)(d + bk) * 2048 + j0 + bj;
        __builtin_prefetch((const void*)(wbase + (size_t)K_TILE * 2048), 0, 1);
#pragma unroll
        for (int g = 0; g < 4; ++g) {
            const float* p = wbase + g * NG;
            bReg[g][0] = p[0];
            bReg[g][1] = p[2048];
        }
    };

    auto storeTiles = [&](int buf) {
        // A: 16 fp32 -> 16 bf16 via native converts, two 16B LDS stores
        v16f af;
        af.s0 = aReg[0].x; af.s1 = aReg[0].y; af.s2 = aReg[0].z; af.s3 = aReg[0].w;
        af.s4 = aReg[1].x; af.s5 = aReg[1].y; af.s6 = aReg[1].z; af.s7 = aReg[1].w;
        af.s8 = aReg[2].x; af.s9 = aReg[2].y; af.sa = aReg[2].z; af.sb = aReg[2].w;
        af.sc = aReg[3].x; af.sd = aReg[3].y; af.se = aReg[3].z; af.sf = aReg[3].w;
        v16bf ab = __builtin_convertvector(af, v16bf);
        v8bf alo = __builtin_shufflevector(ab, ab, 0,1,2,3,4,5,6,7);
        v8bf ahi = __builtin_shufflevector(ab, ab, 8,9,10,11,12,13,14,15);
        *(v8bf*)&As[buf][arow][ahalf * 16]     = alo;   // 16B aligned (row stride 80B)
        *(v8bf*)&As[buf][arow][ahalf * 16 + 8] = ahi;
        // B: j-major (transposed) so B-fragment reads are contiguous per lane
#pragma unroll
        for (int g = 0; g < 4; ++g) {
            v2f t; t.x = bReg[g][0]; t.y = bReg[g][1];
            *(v2bf*)&Bs[buf][g][bj][bk] = __builtin_convertvector(t, v2bf);
        }
    };

    v8f acc[2][4] = {};   // [j-subtile 0/1][gate i,f,g,o], fp32 accumulate

#pragma unroll 1
    for (int kt = 0; kt < K_ITERS; ++kt) {
        const int buf = kt & 1;
        storeTiles(buf);
        __syncthreads();
        if (kt + 1 < K_ITERS) { loadA(kt + 1); loadB(kt + 1); }  // overlap next tile

        // ---- issue ALL fragment loads first (one dscnt wait), then 8 WMMAs ----
        // A fragment: lane holds row M=lane16; K 16*hi..16*hi+15 contiguous
        const int ar = wave * 16 + lane16;
        v8bf a_lo = *(const v8bf*)&As[buf][ar][16 * hi];
        v8bf a_hi = *(const v8bf*)&As[buf][ar][16 * hi + 8];
        // B fragments: lane = column N, K 16*hi..16*hi+15 contiguous
        v8bf blo[8], bhi[8];
#pragma unroll
        for (int s = 0; s < 2; ++s) {
            const int br = s * 16 + lane16;
#pragma unroll
            for (int g = 0; g < 4; ++g) {
                const int i = s * 4 + g;
                blo[i] = *(const v8bf*)&Bs[buf][g][br][16 * hi];
                bhi[i] = *(const v8bf*)&Bs[buf][g][br][16 * hi + 8];
            }
        }
        v16bf afrag = __builtin_shufflevector(a_lo, a_hi,
            0,1,2,3,4,5,6,7,8,9,10,11,12,13,14,15);
#pragma unroll
        for (int i = 0; i < 8; ++i) {
            v16bf bfrag = __builtin_shufflevector(blo[i], bhi[i],
                0,1,2,3,4,5,6,7,8,9,10,11,12,13,14,15);
            acc[i >> 2][i & 3] = __builtin_amdgcn_wmma_f32_16x16x32_bf16(
                false, afrag, false, bfrag, (short)0, acc[i >> 2][i & 3],
                false, false);
        }
        __syncthreads();
    }

    // ---- fused LSTM epilogue (wave-local: all 4 gates in this wave's accs) ----
    // D layout: VGPR v -> M = v + 8*hi, lane16 -> N
#pragma unroll
    for (int s = 0; s < 2; ++s) {
        const int jj = j0 + s * 16 + lane16;            // 0..511 within shard
#pragma unroll
        for (int v = 0; v < 8; ++v) {
            const int row = b0 + wave * 16 + v + 8 * hi;
            float iv = fast_sigmoid(acc[s][0][v]);
            float fv = fast_sigmoid(acc[s][1][v]);
            float gv = fast_tanh   (acc[s][2][v]);
            float ov = fast_sigmoid(acc[s][3][v]);
            float cp = cs[((size_t)n * BATCH + row) * NG + jj];
            float c  = fv * cp + iv * gv;
            float h  = ov * fast_tanh(c);
            const size_t o = (size_t)row * UNITS + n * NG + jj;
            xout[o] = h;
            cout[o] = c;
        }
    }
}

extern "C" void kernel_launch(void* const* d_in, const int* in_sizes, int n_in,
                              void* d_out, int out_size, void* d_ws, size_t ws_size,
                              hipStream_t stream) {
    const float* x  = (const float*)d_in[0];
    const float* hs = (const float*)d_in[1];
    const float* cs = (const float*)d_in[2];
    const float* ws = (const float*)d_in[3];
    float* xout = (float*)d_out;
    float* cout = xout + (size_t)BATCH * UNITS;

    dim3 grid(BATCH / M_TILE, NG / J_TILE, NSP);   // (2, 16, 4) = 128 blocks
    dim3 block(THREADS);
    hipLaunchKernelGGL(lstm_cell_wmma, grid, block, 0, stream,
                       x, hs, cs, ws, xout, cout);
}
